// CAI_35141422416140
// MI455X (gfx1250) — matrix operand order repaired and checked by reading, not compiled, not run on hardware
//
#include <hip/hip_runtime.h>
#include <hip/hip_bf16.h>

// ---------------------------------------------------------------------------
// Co-attention kernel for MI455X (gfx1250, wave32, WMMA).
//   cosc = rowsoftmax(A^T) @ SC ;  cofc = rowsoftmax(A) @ FC
//   A = (SC @ W^T) @ FC^T  per batch, N=90 (padded to 96), D=32.
// Roofline: ~14.3 GFLOP vs ~377 MB HBM traffic -> bandwidth bound with
// bf16 V_WMMA_F32_16X16X32_BF16 (K=32 matches D exactly).
// One workgroup (192 threads = 6 wave32) per batch; all tiles staged in LDS.
// ---------------------------------------------------------------------------

#define NUM_NODE 90
#define FEAT     32
#define NPAD     96
#define BATCH_ELEMS (NUM_NODE * FEAT) /* 2880 */

typedef __bf16 bf16_t;
typedef __attribute__((ext_vector_type(16))) __bf16 v16bf;
typedef __attribute__((ext_vector_type(8)))  __bf16 v8bf;
typedef __attribute__((ext_vector_type(8)))  float  v8f;

// LDS carve (bytes). All offsets multiples of 16 for b128 LDS ops.
#define OFF_SC   0        /* bf16 [96][40]  row-major SC          7680 B */
#define OFF_FC   7680     /* bf16 [96][40]  row-major FC          7680 B */
#define OFF_W    15360    /* bf16 [32][40]  row-major W (e,k)     2560 B */
#define OFF_SCT  17920    /* bf16 [32][104] SC transposed (d,n)   6656 B */
#define OFF_FCT  24576    /* bf16 [32][104] FC transposed (d,n)   6656 B */
#define OFF_EC   31232    /* bf16 [96][40]  EC = SC @ W^T         7680 B */
#define OFF_A    38912    /* f32  [96][100] A = EC @ FC^T        38400 B */
#define OFF_R    77312    /* bf16 [96][104] rowsoftmax(A)        19968 B */
#define OFF_RT   97280    /* bf16 [96][104] rowsoftmax(A^T)      19968 B */
#define SMEM_BYTES 117248

// A-operand gather (16x32 bf16 tile): lane(0-15)=row M, K 0..7 & 16..23;
// lane(16-31)=same row, K 8..15 & 24..31.  Two ds_load_b128 per lane.
__device__ __forceinline__ v16bf ld_a(const bf16_t* P, int ld, int r0, int k0,
                                      int lr, int half) {
  const bf16_t* row = P + (r0 + lr) * ld + k0;
  v8bf lo = *(const v8bf*)(row + half * 8);
  v8bf hi = *(const v8bf*)(row + 16 + half * 8);
  v16bf a;
#pragma unroll
  for (int i = 0; i < 8; ++i) { a[i] = lo[i]; a[8 + i] = hi[i]; }
  return a;
}

// B-operand gather (32x16 bf16 tile) from a transposed (N-major) buffer:
// lane(0-15)=col N, K 0..7 & 16..23 ; lane(16-31)=col N, K 8..15 & 24..31.
__device__ __forceinline__ v16bf ld_b(const bf16_t* Pt, int ld, int c0, int k0,
                                      int lr, int half) {
  const bf16_t* row = Pt + (c0 + lr) * ld + k0;
  v8bf lo = *(const v8bf*)(row + half * 8);
  v8bf hi = *(const v8bf*)(row + 16 + half * 8);
  v16bf v;
#pragma unroll
  for (int i = 0; i < 8; ++i) { v[i] = lo[i]; v[8 + i] = hi[i]; }
  return v;
}

__device__ __forceinline__ v8f wmma_bf16(v16bf a, v16bf b, v8f c) {
  return __builtin_amdgcn_wmma_f32_16x16x32_bf16(
      /*neg_a=*/false, a, /*neg_b=*/false, b,
      /*c_mod=*/(short)0, c, /*reuse_a=*/false, /*reuse_b=*/false);
}

__global__ void __launch_bounds__(192)
coattn_kernel(const float* __restrict__ sc_g, const float* __restrict__ fc_g,
              const float* __restrict__ w_g, float* __restrict__ cosc,
              float* __restrict__ cofc) {
  extern __shared__ char smem[];
  bf16_t* sSC  = (bf16_t*)(smem + OFF_SC);
  bf16_t* sFC  = (bf16_t*)(smem + OFF_FC);
  bf16_t* sW   = (bf16_t*)(smem + OFF_W);
  bf16_t* sSCt = (bf16_t*)(smem + OFF_SCT);
  bf16_t* sFCt = (bf16_t*)(smem + OFF_FCT);
  bf16_t* sEC  = (bf16_t*)(smem + OFF_EC);
  float*  sA   = (float*) (smem + OFF_A);
  bf16_t* sR   = (bf16_t*)(smem + OFF_R);
  bf16_t* sRt  = (bf16_t*)(smem + OFF_RT);

  const int tid  = threadIdx.x;
  const int wave = tid >> 5;         // 0..5 : row-tile owner
  const int lane = tid & 31;
  const int lr   = lane & 15;
  const int half = lane >> 4;
  const int b    = blockIdx.x;

  // ---- Phase 0: global -> LDS, f32 -> bf16, plus transposed copies -------
  const float4* sc4 = (const float4*)(sc_g + (size_t)b * BATCH_ELEMS);
  const float4* fc4 = (const float4*)(fc_g + (size_t)b * BATCH_ELEMS);
  for (int i = tid; i < BATCH_ELEMS / 4; i += 192) {
    float4 v = sc4[i];
    float4 u = fc4[i];
    int n = i >> 3;
    int d = (i & 7) << 2;
    sSC[n * 40 + d + 0] = (bf16_t)v.x; sSC[n * 40 + d + 1] = (bf16_t)v.y;
    sSC[n * 40 + d + 2] = (bf16_t)v.z; sSC[n * 40 + d + 3] = (bf16_t)v.w;
    sFC[n * 40 + d + 0] = (bf16_t)u.x; sFC[n * 40 + d + 1] = (bf16_t)u.y;
    sFC[n * 40 + d + 2] = (bf16_t)u.z; sFC[n * 40 + d + 3] = (bf16_t)u.w;
    sSCt[(d + 0) * 104 + n] = (bf16_t)v.x; sSCt[(d + 1) * 104 + n] = (bf16_t)v.y;
    sSCt[(d + 2) * 104 + n] = (bf16_t)v.z; sSCt[(d + 3) * 104 + n] = (bf16_t)v.w;
    sFCt[(d + 0) * 104 + n] = (bf16_t)u.x; sFCt[(d + 1) * 104 + n] = (bf16_t)u.y;
    sFCt[(d + 2) * 104 + n] = (bf16_t)u.z; sFCt[(d + 3) * 104 + n] = (bf16_t)u.w;
  }
  const float4* w4 = (const float4*)w_g;
  for (int i = tid; i < 256; i += 192) {
    float4 v = w4[i];
    int e = i >> 3;
    int k = (i & 7) << 2;
    sW[e * 40 + k + 0] = (bf16_t)v.x; sW[e * 40 + k + 1] = (bf16_t)v.y;
    sW[e * 40 + k + 2] = (bf16_t)v.z; sW[e * 40 + k + 3] = (bf16_t)v.w;
  }
  {  // zero-pad rows 90..95 (row-major) and cols 90..95 (transposed), 192 ea.
    int n = 90 + (tid >> 5), d = tid & 31;
    sSC[n * 40 + d] = (bf16_t)0.f;
    sFC[n * 40 + d] = (bf16_t)0.f;
    int dd = tid / 6, nn = 90 + (tid % 6);
    sSCt[dd * 104 + nn] = (bf16_t)0.f;
    sFCt[dd * 104 + nn] = (bf16_t)0.f;
  }
  __syncthreads();

  const int r0 = wave * 16;  // this wave's row-tile base

  // ---- Phase 1: EC = SC @ W^T   ([96x32], K=32, 2 WMMA per wave) ---------
  {
    v16bf aop = ld_a(sSC, 40, r0, 0, lr, half);
#pragma unroll
    for (int et = 0; et < 2; ++et) {
      v16bf bop = ld_b(sW, 40, et * 16, 0, lr, half);
      v8f acc = {};
      acc = wmma_bf16(aop, bop, acc);
#pragma unroll
      for (int j = 0; j < 8; ++j)  // C/D layout: lane half selects M +8
        sEC[(r0 + half * 8 + j) * 40 + (et * 16 + lr)] = (bf16_t)acc[j];
    }
  }
  __syncthreads();

  // ---- Phase 2: A = EC @ FC^T   ([96x96], K=32, 6 WMMA per wave) ---------
  {
    v16bf aop = ld_a(sEC, 40, r0, 0, lr, half);
#pragma unroll
    for (int mt = 0; mt < 6; ++mt) {
      v16bf bop = ld_b(sFC, 40, mt * 16, 0, lr, half);
      v8f acc = {};
      acc = wmma_bf16(aop, bop, acc);
#pragma unroll
      for (int j = 0; j < 8; ++j)
        sA[(r0 + half * 8 + j) * 100 + (mt * 16 + lr)] = acc[j];
    }
  }
  __syncthreads();

  // ---- Phase 3: softmaxes. threads 0-95: R=rowsoftmax(A) (row reads);
  //               threads 96-191: Rt=rowsoftmax(A^T) (conflict-free col reads)
  if (tid < 96) {
    const int r = tid;
    float mx = -3.0e38f;
    for (int n = 0; n < NUM_NODE; ++n) mx = fmaxf(mx, sA[r * 100 + n]);
    float s = 0.f;
    for (int n = 0; n < NUM_NODE; ++n) s += __expf(sA[r * 100 + n] - mx);
    float inv = __frcp_rn(s);
    for (int n = 0; n < NUM_NODE; ++n)
      sR[r * 104 + n] = (bf16_t)(__expf(sA[r * 100 + n] - mx) * inv);
    for (int n = NUM_NODE; n < NPAD; ++n) sR[r * 104 + n] = (bf16_t)0.f;
  } else {
    const int r = tid - 96;
    float mx = -3.0e38f;
    for (int n = 0; n < NUM_NODE; ++n) mx = fmaxf(mx, sA[n * 100 + r]);
    float s = 0.f;
    for (int n = 0; n < NUM_NODE; ++n) s += __expf(sA[n * 100 + r] - mx);
    float inv = __frcp_rn(s);
    for (int n = 0; n < NUM_NODE; ++n)
      sRt[r * 104 + n] = (bf16_t)(__expf(sA[n * 100 + r] - mx) * inv);
    for (int n = NUM_NODE; n < NPAD; ++n) sRt[r * 104 + n] = (bf16_t)0.f;
  }
  __syncthreads();

  // ---- Phase 4: cosc = Rt @ SC, cofc = R @ FC (K=96 -> 3 chunks of 32) ---
#pragma unroll
  for (int dt = 0; dt < 2; ++dt) {
    v8f acc_q = {};  // cosc tile
    v8f acc_e = {};  // cofc tile
#pragma unroll
    for (int kt = 0; kt < 3; ++kt) {
      v16bf aQ = ld_a(sRt, 104, r0, kt * 32, lr, half);
      v16bf aE = ld_a(sR,  104, r0, kt * 32, lr, half);
      v16bf bS = ld_b(sSCt, 104, dt * 16, kt * 32, lr, half);
      v16bf bF = ld_b(sFCt, 104, dt * 16, kt * 32, lr, half);
      acc_q = wmma_bf16(aQ, bS, acc_q);
      acc_e = wmma_bf16(aE, bF, acc_e);
    }
#pragma unroll
    for (int j = 0; j < 8; ++j) {
      int m = r0 + half * 8 + j;
      if (m < NUM_NODE) {
        int d = dt * 16 + lr;
        size_t o = ((size_t)b * NUM_NODE + m) * FEAT + d;
        cosc[o] = acc_q[j];
        cofc[o] = acc_e[j];
      }
    }
  }
}

extern "C" void kernel_launch(void* const* d_in, const int* in_sizes, int n_in,
                              void* d_out, int out_size, void* d_ws, size_t ws_size,
                              hipStream_t stream) {
  (void)n_in; (void)d_ws; (void)ws_size;
  const float* sc = (const float*)d_in[0];
  const float* fc = (const float*)d_in[1];
  const float* We = (const float*)d_in[2];
  float* out = (float*)d_out;

  const int batch = in_sizes[0] / BATCH_ELEMS;  // 8192
  float* cosc = out;
  float* cofc = out + (size_t)batch * BATCH_ELEMS;  // second tuple output

  coattn_kernel<<<batch, 192, SMEM_BYTES, stream>>>(sc, fc, We, cosc, cofc);
  (void)out_size;
}
